// QLinear_30588757082117
// MI455X (gfx1250) — 8-bit matrix operand order repaired and checked by reading, compiled, not run on hardware
//
#include <hip/hip_runtime.h>
#include <hip/hip_bf16.h>
#include <stdint.h>

#define MDIM 8192
#define KDIM 4096
#define NDIM 4096
#define EPSQ 1e-7f

typedef __attribute__((ext_vector_type(8))) int v8i;

union AFrag { v8i v; long d[4]; };
union BFrag { v8i v; int4 q[2]; };

__global__ __launch_bounds__(256) void quant_rows_kernel(
    const float* __restrict__ x, int8_t* __restrict__ xq,
    float* __restrict__ xs) {
  const int row = blockIdx.x;
  const float* xr = x + (size_t)row * KDIM;
  __shared__ float red[256];

  float amax = 0.0f;
  const float4* xr4 = (const float4*)xr;
  #pragma unroll
  for (int it = 0; it < 4; ++it) {
    float4 v = xr4[threadIdx.x + it * 256];
    amax = fmaxf(amax, fabsf(v.x));
    amax = fmaxf(amax, fabsf(v.y));
    amax = fmaxf(amax, fabsf(v.z));
    amax = fmaxf(amax, fabsf(v.w));
  }
  red[threadIdx.x] = amax;
  __syncthreads();
  #pragma unroll
  for (int s = 128; s > 0; s >>= 1) {
    if ((int)threadIdx.x < s)
      red[threadIdx.x] = fmaxf(red[threadIdx.x], red[threadIdx.x + s]);
    __syncthreads();
  }
  const float scale = fmaxf(red[0], EPSQ) / 127.0f;
  const float inv = 1.0f / scale;
  if (threadIdx.x == 0) xs[row] = scale;

  char4* out4 = (char4*)(xq + (size_t)row * KDIM);
  #pragma unroll
  for (int it = 0; it < 4; ++it) {
    float4 v = xr4[threadIdx.x + it * 256];
    float qx = fminf(fmaxf(rintf(v.x * inv), -128.0f), 127.0f);
    float qy = fminf(fmaxf(rintf(v.y * inv), -128.0f), 127.0f);
    float qz = fminf(fmaxf(rintf(v.z * inv), -128.0f), 127.0f);
    float qw = fminf(fmaxf(rintf(v.w * inv), -128.0f), 127.0f);
    char4 c;
    c.x = (int8_t)(int)qx; c.y = (int8_t)(int)qy;
    c.z = (int8_t)(int)qz; c.w = (int8_t)(int)qw;
    out4[threadIdx.x + it * 256] = c;
  }
}

__global__ __launch_bounds__(256) void gemm_i8_wmma_kernel(
    const int8_t* __restrict__ xq, const int8_t* __restrict__ wq,
    const float* __restrict__ xs, const float* __restrict__ wsc,
    const _Float16* __restrict__ bias, _Float16* __restrict__ out) {
  const int lane = threadIdx.x & 31;
  const int wave = threadIdx.x >> 5;
  const int waveM = wave >> 1;
  const int waveN = wave & 1;

  const int m0 = blockIdx.x * 128 + waveM * 32;
  const int n0 = blockIdx.y * 128 + waveN * 64;

  const int l15 = lane & 15;
  const int akoff = (lane < 16) ? 0 : 16;
  const int bkoff = (lane < 16) ? 0 : 16;

  const int8_t* aBase = xq + (size_t)(m0 + l15) * KDIM + akoff;
  const int8_t* bBase = wq + (size_t)(n0 + l15) * KDIM + bkoff;

  v8i acc[2][4] = {};

  for (int k = 0; k < KDIM; k += 64) {
    AFrag a[2];
    BFrag b[4];
    #pragma unroll
    for (int i = 0; i < 2; ++i) {
      const int8_t* ap = aBase + (size_t)(i * 16) * KDIM + k;
      #pragma unroll
      for (int t = 0; t < 4; ++t) a[i].d[t] = *(const long*)(ap + (t & 1) * 8 + (t >> 1) * 32);
    }
    #pragma unroll
    for (int j = 0; j < 4; ++j) {
      const int8_t* bp = bBase + (size_t)(j * 16) * KDIM + k;
      b[j].q[0] = *(const int4*)(bp);
      b[j].q[1] = *(const int4*)(bp + 32);
    }

    #pragma unroll
    for (int i = 0; i < 2; ++i)
      #pragma unroll
      for (int j = 0; j < 4; ++j)
        acc[i][j] = __builtin_amdgcn_wmma_i32_16x16x64_iu8(
             true, a[i].v,  true, b[j].v, acc[i][j],
             false,  false);
  }

  const int rbase = (lane >= 16) ? 8 : 0;
  #pragma unroll
  for (int j = 0; j < 4; ++j) {
    const int coln = n0 + j * 16 + l15;
    const float wsn = wsc[coln];
    const float bn = (float)bias[coln];
    #pragma unroll
    for (int i = 0; i < 2; ++i) {
      #pragma unroll
      for (int r = 0; r < 8; ++r) {
        const int rowm = m0 + i * 16 + rbase + r;
        const float v = (float)acc[i][j][r] * xs[rowm] * wsn + bn;
        out[(size_t)rowm * NDIM + coln] = (_Float16)v;
      }
    }
  }
}

extern "C" void kernel_launch(void* const* d_in, const int* in_sizes, int n_in,
                              void* d_out, int out_size, void* d_ws,
                              size_t ws_size, hipStream_t stream) {
  (void)in_sizes; (void)n_in; (void)out_size; (void)ws_size;
  const float* x = (const float*)d_in[0];
  const int8_t* wq = (const int8_t*)d_in[1];
  const float* wsc = (const float*)d_in[2];
  const _Float16* bias = (const _Float16*)d_in[3];
  _Float16* out = (_Float16*)d_out;

  int8_t* xq = (int8_t*)d_ws;
  float* xs = (float*)((char*)d_ws + (size_t)MDIM * KDIM);

  quant_rows_kernel<<<MDIM, 256, 0, stream>>>(x, xq, xs);

  dim3 grid(MDIM / 128, NDIM / 128);
  gemm_i8_wmma_kernel<<<grid, 256, 0, stream>>>(xq, wq, xs, wsc, bias, out);
}
